// SO3onS2_45758581572036
// MI455X (gfx1250) — matrix operand order repaired and checked by reading, not compiled, not run on hardware
//
#include <hip/hip_runtime.h>

// ---------------------------------------------------------------------------
// SO3-on-S2: SHT (fused DFT x Legendre as one bf16 WMMA GEMM) + CG couplings
// (f32 WMMA 16x16x4 batched GEMMs). Target: MI455X / gfx1250, wave32.
// ---------------------------------------------------------------------------

typedef __attribute__((ext_vector_type(16))) __bf16 v16bf;
typedef __attribute__((ext_vector_type(8)))  float  v8f;
typedef __attribute__((ext_vector_type(2)))  float  v2f;

#define LMAXC   8
#define NPAIR   45
#define NLATC   128
#define NLONC   256
#define BATCHC  2048
#define KTOT    32768            // NLAT*NLON
#define NCOL    96               // 45 (l,m) pairs * {re,im} = 90, padded to 96
#define NOUTC   215              // total coupled outputs per batch
#define KSPLIT  16
#define NCHUNK  1024             // KTOT / 32
#define CELEMS  196608           // BATCH * NCOL
#define KPAD    304              // max n1*n2 (=289) padded to 16

__device__ __forceinline__ unsigned short f2bf(float x) {
  unsigned int u = __float_as_uint(x);
  u += 0x7FFFu + ((u >> 16) & 1u);          // round-to-nearest-even
  return (unsigned short)(u >> 16);
}

// ---------------------------------------------------------------------------
// Kernel 1: Clenshaw-Curtis weights + normalized assoc. Legendre -> Wtab[45][128]
// ---------------------------------------------------------------------------
__global__ void sht_legendre_k(float* __restrict__ Wtab) {
  int k = threadIdx.x;            // 0..127
  if (k >= NLATC) return;
  double theta = 3.14159265358979323846 * (double)k / 127.0;
  double x = cos(theta);
  double st2 = 1.0 - x * x;
  double sint = st2 > 0.0 ? sqrt(st2) : 0.0;
  double s = 0.0;
  for (int j = 1; j <= 63; ++j)
    s += (2.0 / (4.0 * j * j - 1.0)) * cos(2.0 * theta * j);
  double w = (2.0 / 127.0) * (1.0 - s);
  if (k == 0 || k == 127) w *= 0.5;

  double p[9][9];
  p[0][0] = sqrt(1.0 / (4.0 * 3.14159265358979323846));
  for (int m = 1; m <= LMAXC; ++m)
    p[m][m] = -sqrt((2.0 * m + 1.0) / (2.0 * m)) * sint * p[m - 1][m - 1];
  for (int m = 0; m <= LMAXC; ++m) {
    if (m + 1 <= LMAXC)
      p[m][m + 1] = sqrt(2.0 * m + 3.0) * x * p[m][m];
    for (int l = m + 2; l <= LMAXC; ++l) {
      double a = sqrt((4.0 * l * l - 1.0) / ((double)l * l - (double)m * m));
      double b = sqrt(((l - 1.0) * (l - 1.0) - (double)m * m) /
                      (4.0 * (l - 1.0) * (l - 1.0) - 1.0));
      p[m][l] = a * (x * p[m][l - 1] - b * p[m][l - 2]);
    }
  }
  for (int l = 0; l <= LMAXC; ++l)
    for (int m = 0; m <= l; ++m) {
      int pi = l * (l + 1) / 2 + m;
      Wtab[pi * NLATC + k] = (float)(p[m][l] * w);
    }
}

// ---------------------------------------------------------------------------
// Kernel 2: bf16 projection matrix, pre-swizzled in WMMA B-fragment order.
//   Bmat[chunk][ntile][lane][e], lane: n = lane&15, h = lane>>4,
//   element e: K-in-chunk = h*16 + e  (contiguous k order, same as the A frag).
// ---------------------------------------------------------------------------
__global__ void sht_projmat_k(const float* __restrict__ Wtab,
                              unsigned short* __restrict__ Bmat) {
  long long idx = (long long)blockIdx.x * 256 + threadIdx.x;
  const long long total = (long long)NCHUNK * 6 * 32 * 16;
  if (idx >= total) return;
  int e = (int)(idx & 15);
  long long t2 = idx >> 4;
  int lane = (int)(t2 & 31);
  long long t3 = t2 >> 5;
  int tile = (int)(t3 % 6);
  int chunk = (int)(t3 / 6);

  int ncol = tile * 16 + (lane & 15);
  int h = lane >> 4;
  int kin = h * 16 + e;
  int kn = chunk * 32 + kin;
  int klat = kn >> 8;
  int nlon = kn & 255;

  float val = 0.0f;
  if (ncol < 2 * NPAIR) {
    int pairIdx = ncol >> 1;
    int reim = ncol & 1;
    int l = 0;
    while ((l + 1) * (l + 2) / 2 <= pairIdx) ++l;
    int m = pairIdx - l * (l + 1) / 2;
    float ang = (6.28318530717958647692f / (float)NLONC) * (float)(m * nlon);
    float base = (6.28318530717958647692f / (float)NLONC) *
                 Wtab[pairIdx * NLATC + klat];
    val = (reim == 0) ? base * __cosf(ang) : -base * __sinf(ang);
  }
  Bmat[idx] = f2bf(val);
}

// ---------------------------------------------------------------------------
// Kernel 3: Clebsch-Gordan matrices, packed back-to-back in pair order.
// ---------------------------------------------------------------------------
__global__ void cg_build_k(float* __restrict__ cgout, long long total) {
  long long gid = (long long)blockIdx.x * 256 + threadIdx.x;
  if (gid >= total) return;
  long long rem = gid;
  int l1 = -1, l2 = -1;
  for (int a = 0; a <= LMAXC; ++a)
    for (int b = a; b <= LMAXC; ++b) {
      if (l1 < 0) {
        long long nn = (long long)(2 * a + 1) * (2 * b + 1);
        long long sz = nn * nn;
        if (rem < sz) { l1 = a; l2 = b; }
        else rem -= sz;
      }
    }
  if (l1 < 0) return;
  int n2 = 2 * l2 + 1;
  int K = (2 * l1 + 1) * n2;
  int row = (int)(rem / K);
  int col = (int)(rem % K);
  int m1 = (row / n2) - l1;
  int m2 = (row % n2) - l2;
  int l = l2 - l1, cc = col;
  while (cc >= 2 * l + 1) { cc -= 2 * l + 1; ++l; }
  int m = cc - l;

  double F[34];
  F[0] = 1.0;
  for (int q = 1; q < 34; ++q) F[q] = F[q - 1] * (double)q;

  double v = 0.0;
  if (m1 + m2 == m) {
    double pre = sqrt((2.0 * l + 1.0) * F[l1 + l2 - l] * F[l1 - l2 + l] *
                      F[-l1 + l2 + l] / F[l1 + l2 + l + 1]);
    pre *= sqrt(F[l + m] * F[l - m] * F[l1 - m1] * F[l1 + m1] *
                F[l2 - m2] * F[l2 + m2]);
    double s = 0.0;
    for (int k = 0; k <= l1 + l2 - l; ++k) {
      int d0 = k, d1 = l1 + l2 - l - k, d2 = l1 - m1 - k;
      int d3 = l2 + m2 - k, d4 = l - l2 + m1 + k, d5 = l - l1 - m2 + k;
      if (d0 < 0 || d1 < 0 || d2 < 0 || d3 < 0 || d4 < 0 || d5 < 0) continue;
      double den = F[d0] * F[d1] * F[d2] * F[d3] * F[d4] * F[d5];
      s += ((k & 1) ? -1.0 : 1.0) / den;
    }
    v = pre * s;
  }
  cgout[gid] = (float)v;
}

// ---------------------------------------------------------------------------
// Kernel 4: main GEMM (split-K), f(2048x32768) x C^T via bf16 WMMA.
// ---------------------------------------------------------------------------
__global__ __launch_bounds__(256) void sht_gemm_k(
    const float* __restrict__ f, const unsigned short* __restrict__ Bmat,
    float* __restrict__ partial) {
  __shared__ unsigned short As[128 * 40];
  __shared__ unsigned short Bs[6 * 32 * 16];

  const int t = threadIdx.x;
  const int mbase = blockIdx.x * 128;
  const int ks = blockIdx.y;
  const int wv = t >> 5;
  const int L = t & 31;
  const int h = L >> 4;
  const int ln = L & 15;

  v8f acc[6];
#pragma unroll
  for (int nt = 0; nt < 6; ++nt) acc[nt] = (v8f){0.f,0.f,0.f,0.f,0.f,0.f,0.f,0.f};

  const int arow = t >> 1;
  const int ahalf = t & 1;
  const float* asrc = f + (size_t)(mbase + arow) * KTOT + ahalf * 16;

  for (int cc2 = 0; cc2 < NCHUNK / KSPLIT; ++cc2) {
    const int chunk = ks * (NCHUNK / KSPLIT) + cc2;
    const size_t kbase = (size_t)chunk * 32;

    // prefetch next chunk's A slice (speculative, hides latency)
    __builtin_prefetch(asrc + kbase + 32, 0, 1);

    const unsigned int* bsrc =
        (const unsigned int*)(Bmat + (size_t)chunk * 3072);
    unsigned int* bdst = (unsigned int*)Bs;
#pragma unroll
    for (int i = 0; i < 6; ++i) bdst[t * 6 + i] = bsrc[t * 6 + i];

    const float4* ap = (const float4*)(asrc + kbase);
    float4 v0 = ap[0], v1 = ap[1], v2 = ap[2], v3 = ap[3];
    __align__(16) unsigned short tmp[16];
    tmp[0] = f2bf(v0.x); tmp[1] = f2bf(v0.y); tmp[2] = f2bf(v0.z); tmp[3] = f2bf(v0.w);
    tmp[4] = f2bf(v1.x); tmp[5] = f2bf(v1.y); tmp[6] = f2bf(v1.z); tmp[7] = f2bf(v1.w);
    tmp[8] = f2bf(v2.x); tmp[9] = f2bf(v2.y); tmp[10]= f2bf(v2.z); tmp[11]= f2bf(v2.w);
    tmp[12]= f2bf(v3.x); tmp[13]= f2bf(v3.y); tmp[14]= f2bf(v3.z); tmp[15]= f2bf(v3.w);
    unsigned short* adst = As + arow * 40 + ahalf * 16;
    ((uint4*)adst)[0] = ((uint4*)tmp)[0];
    ((uint4*)adst)[1] = ((uint4*)tmp)[1];

    __syncthreads();

    const unsigned short* al = As + (wv * 16 + ln) * 40;
    v16bf afrag;
    ((uint4*)&afrag)[0] = *(const uint4*)(al + h * 16);
    ((uint4*)&afrag)[1] = *(const uint4*)(al + h * 16 + 8);

#pragma unroll
    for (int nt = 0; nt < 6; ++nt) {
      const uint4* bp = (const uint4*)(Bs + (nt * 32 + L) * 16);
      v16bf bfrag;
      ((uint4*)&bfrag)[0] = bp[0];
      ((uint4*)&bfrag)[1] = bp[1];
      acc[nt] = __builtin_amdgcn_wmma_f32_16x16x32_bf16(
          false, afrag, false, bfrag, (short)0, acc[nt], false, false);
    }
    __syncthreads();
  }

  float* pbase = partial + (size_t)ks * BATCHC * NCOL;
#pragma unroll
  for (int nt = 0; nt < 6; ++nt)
#pragma unroll
    for (int v = 0; v < 8; ++v) {
      int M = h * 8 + v;
      pbase[(size_t)(mbase + wv * 16 + M) * NCOL + nt * 16 + ln] = acc[nt][v];
    }
}

// ---------------------------------------------------------------------------
// Kernel 5: split-K reduction -> coeffs[b][96]
// ---------------------------------------------------------------------------
__global__ void sht_reduce_k(const float* __restrict__ partial,
                             float* __restrict__ coeffs) {
  int i = blockIdx.x * 256 + threadIdx.x;
  if (i >= CELEMS) return;
  float s = 0.f;
#pragma unroll
  for (int k = 0; k < KSPLIT; ++k) s += partial[(size_t)k * CELEMS + i];
  coeffs[i] = s;
}

// ---------------------------------------------------------------------------
// Kernel 6: CG couplings via f32 WMMA 16x16x4.
// One block = (pair p, tile of 16 batches). TR = TP @ CG computed with WMMA:
//   A = TP (M=batch 16, K up to 289), B = CG rows from L2, D = TR.
// A-frag (32-bit 16x4): lane L -> M=L&15, elem v -> K = (L>>4)*2 + v.
// D-frag (16x16 f32):   lane L, vgpr v -> M = v + 8*(L>>4), N = L&15.
// ---------------------------------------------------------------------------
__global__ __launch_bounds__(256) void cg_couple_k(
    const float* __restrict__ coeffs, const float* __restrict__ cg,
    float* __restrict__ out) {
  const int p = blockIdx.x;
  const int btile = blockIdx.y;           // 0..127 (16 batches each)
  const int t = threadIdx.x;
  const int wv = t >> 5;
  const int L = t & 31;
  const int h = L >> 4;
  const int ln = L & 15;

  __shared__ float fcre[16 * 81], fcim[16 * 81];
  __shared__ float tpre[16 * KPAD], tpim[16 * KPAD];   // TP, then aliased as TR

  // decode pair p -> (l1,l2); cg offset and output base
  int l1 = 0, l2 = 0;
  size_t cgb = 0;
  int ob = 0;
  {
    int c = 0;
    for (int a = 0; a <= LMAXC; ++a)
      for (int bb = a; bb <= LMAXC; ++bb) {
        if (c == p) { l1 = a; l2 = bb; }
        if (c < p) {
          long long nn = (long long)(2 * a + 1) * (2 * bb + 1);
          cgb += (size_t)(nn * nn);
          for (int l = bb - a; l <= a + bb; ++l)
            if (l <= LMAXC) ++ob;
        }
        ++c;
      }
  }
  const int n2 = 2 * l2 + 1;
  const int K = (2 * l1 + 1) * n2;
  const int NT = (K + 15) >> 4;           // N tiles of TR
  const int KT = (K + 3) >> 2;            // K tiles (WMMA K=4)
  const float* cgp = cg + cgb;

  // --- build fc for 16 batches ---
  for (int i = t; i < 16 * 81; i += 256) {
    int bb = i / 81, q = i - bb * 81;
    int l = 0;
    while ((l + 1) * (l + 1) <= q) ++l;
    int m = (q - l * l) - l;
    const float* cb = coeffs + (size_t)(btile * 16 + bb) * NCOL;
    float re, im;
    if (m >= 0) {
      int pi = l * (l + 1) / 2 + m;
      re = cb[2 * pi]; im = cb[2 * pi + 1];
    } else {
      int mm = -m;
      int pi = l * (l + 1) / 2 + mm;
      float sg = (mm & 1) ? -1.f : 1.f;
      re = sg * cb[2 * pi]; im = -sg * cb[2 * pi + 1];
    }
    fcre[i] = re; fcim[i] = im;
  }
  __syncthreads();

  // --- build TP (zero-padded to KPAD so padded A rows kill garbage B) ---
  for (int i = t; i < 16 * KPAD; i += 256) {
    int bb = i / KPAD, k = i - bb * KPAD;
    float re = 0.f, im = 0.f;
    if (k < K) {
      int i1 = k / n2, i2 = k - i1 * n2;
      float ar = fcre[bb * 81 + l1 * l1 + i1], ai = fcim[bb * 81 + l1 * l1 + i1];
      float br = fcre[bb * 81 + l2 * l2 + i2], bi = fcim[bb * 81 + l2 * l2 + i2];
      re = ar * br - ai * bi;
      im = ar * bi + ai * br;
    }
    tpre[i] = re; tpim[i] = im;
  }
  __syncthreads();

  // --- TR = TP @ CG with f32 WMMA; waves split N tiles (<=19 -> <=3 each) ---
  int nts[3];
  int nacc = 0;
  for (int nt = wv; nt < NT; nt += 8) nts[nacc++] = nt;
  v8f accre[3], accim[3];
  for (int a = 0; a < nacc; ++a) {
    accre[a] = (v8f){0.f,0.f,0.f,0.f,0.f,0.f,0.f,0.f};
    accim[a] = (v8f){0.f,0.f,0.f,0.f,0.f,0.f,0.f,0.f};
  }
  for (int kt = 0; kt < KT; ++kt) {
    const int r0 = kt * 4 + h * 2;
    v2f are, aim;
    are[0] = tpre[ln * KPAD + r0];
    are[1] = tpre[ln * KPAD + r0 + 1];
    aim[0] = tpim[ln * KPAD + r0];
    aim[1] = tpim[ln * KPAD + r0 + 1];
    const int r0c = (r0 < K) ? r0 : (K - 1);
    const int r1c = (r0 + 1 < K) ? (r0 + 1) : (K - 1);
    for (int a = 0; a < nacc; ++a) {
      int col = nts[a] * 16 + ln;
      int colc = (col < K) ? col : (K - 1);
      v2f bf;
      bf[0] = cgp[(size_t)r0c * K + colc];
      bf[1] = cgp[(size_t)r1c * K + colc];
      accre[a] = __builtin_amdgcn_wmma_f32_16x16x4_f32(
          false, are, false, bf, (short)0, accre[a], false, false);
      accim[a] = __builtin_amdgcn_wmma_f32_16x16x4_f32(
          false, aim, false, bf, (short)0, accim[a], false, false);
    }
  }
  __syncthreads();           // all waves done reading TP

  // --- dump TR into the TP buffers (alias) ---
  for (int a = 0; a < nacc; ++a) {
    int nt = nts[a];
#pragma unroll
    for (int v = 0; v < 8; ++v) {
      int M = h * 8 + v;                         // batch within tile
      tpre[M * KPAD + nt * 16 + ln] = accre[a][v];
      tpim[M * KPAD + nt * 16 + ln] = accim[a][v];
    }
  }
  __syncthreads();

  // --- project tr segments onto conj(fc[l]) ---
  {
    int bb = t & 15;
    int s = t >> 4;                               // 0..15
    int lmin = l2 - l1;
    int lhi = l1 + l2; if (lhi > LMAXC) lhi = LMAXC;
    int V = lhi - lmin + 1;
    if (s < V) {
      int l = lmin + s;
      int off = l * l - lmin * lmin;              // column offset of segment
      float sr = 0.f, si = 0.f;
      for (int j = 0; j < 2 * l + 1; ++j) {
        float tr_ = tpre[bb * KPAD + off + j], ti_ = tpim[bb * KPAD + off + j];
        float fr = fcre[bb * 81 + l * l + j], fi = fcim[bb * 81 + l * l + j];
        sr += tr_ * fr + ti_ * fi;                // tr * conj(fc)
        si += ti_ * fr - tr_ * fi;
      }
      size_t o = ((size_t)(btile * 16 + bb) * NOUTC + ob + s) * 2;
      out[o] = sr; out[o + 1] = si;
    }
  }
}

// ---------------------------------------------------------------------------
// Host launcher
// ---------------------------------------------------------------------------
extern "C" void kernel_launch(void* const* d_in, const int* in_sizes, int n_in,
                              void* d_out, int out_size, void* d_ws,
                              size_t ws_size, hipStream_t stream) {
  (void)in_sizes; (void)n_in; (void)out_size; (void)ws_size;
  const float* f = (const float*)d_in[0];
  float* out = (float*)d_out;
  char* ws = (char*)d_ws;

  auto al = [](size_t x) { return (x + 255) & ~(size_t)255; };
  size_t oW = 0;
  size_t oB = al(oW + (size_t)NPAIR * NLATC * 4);
  size_t oP = al(oB + (size_t)NCHUNK * 6 * 32 * 16 * 2);
  size_t oC = al(oP + (size_t)KSPLIT * CELEMS * 4);
  size_t oG = al(oC + (size_t)CELEMS * 4);

  float* Wtab = (float*)(ws + oW);
  unsigned short* Bmat = (unsigned short*)(ws + oB);
  float* partial = (float*)(ws + oP);
  float* coeffs = (float*)(ws + oC);
  float* cgmat = (float*)(ws + oG);

  long long totcg = 0;
  for (int a = 0; a <= LMAXC; ++a)
    for (int b = a; b <= LMAXC; ++b) {
      long long nn = (long long)(2 * a + 1) * (2 * b + 1);
      totcg += nn * nn;
    }

  sht_legendre_k<<<1, 128, 0, stream>>>(Wtab);

  long long projTot = (long long)NCHUNK * 6 * 32 * 16;
  sht_projmat_k<<<(unsigned)((projTot + 255) / 256), 256, 0, stream>>>(Wtab, Bmat);

  cg_build_k<<<(unsigned)((totcg + 255) / 256), 256, 0, stream>>>(cgmat, totcg);

  sht_gemm_k<<<dim3(BATCHC / 128, KSPLIT), 256, 0, stream>>>(f, Bmat, partial);

  sht_reduce_k<<<(CELEMS + 255) / 256, 256, 0, stream>>>(partial, coeffs);

  cg_couple_k<<<dim3(NPAIR, BATCHC / 16), 256, 0, stream>>>(coeffs, cgmat, out);
}
